// ActionPrototypeBank_64304250355736
// MI455X (gfx1250) — matrix operand order repaired and checked by reading, not compiled, not run on hardware
//
#include <hip/hip_runtime.h>
#include <hip/hip_bf16.h>
#include <math.h>

typedef __attribute__((ext_vector_type(16))) _Float16 v16h;
typedef __attribute__((ext_vector_type(8)))  _Float16 v8h;
typedef __attribute__((ext_vector_type(8)))  float    v8f;
typedef __attribute__((ext_vector_type(4)))  float    v4f;

#define B_DIM   16384
#define F_DIM   1024
#define P_DIM   256
#define K_DIM   4096

// ---------------------------------------------------------------------------
// A fragment (16-bit 16x32): lane m=lane&15, h=lane>>4 holds one
// contiguous 16-half run: K = 16h..16h+15 (e=0..15), matching packed B.
__device__ __forceinline__ v16h apb_load_afrag(const _Float16* rowp, int h) {
    v8h a0 = *(const v8h*)(rowp + 16 * h);
    v8h a1 = *(const v8h*)(rowp + 16 * h + 8);
    return __builtin_shufflevector(a0, a1, 0, 1, 2, 3, 4, 5, 6, 7,
                                   8, 9, 10, 11, 12, 13, 14, 15);
}

// ---------------------------------------------------------------------------
__global__ __launch_bounds__(256) void apb_zero_accum(float* accum) {
    if (threadIdx.x < 2 && blockIdx.x == 0) accum[threadIdx.x] = 0.0f;
}

// rstd[i] = 1 / max(std[i], 1e-6)
__global__ __launch_bounds__(256) void apb_rstd(const float* __restrict__ stdv,
                                                float* __restrict__ rstd) {
    int i = blockIdx.x * 256 + threadIdx.x;
    rstd[i] = 1.0f / fmaxf(stdv[i], 1e-6f);
}

// Pack pca [F=1024 (k), P=256 (n)] into B-fragment order:
//   idx = ((tn*32 + tk)*32 + lane)*16 + e ;  K = tk*32 + (lane>>4)*16 + e ; N = tn*16 + (lane&15)
__global__ __launch_bounds__(256) void apb_pack_pca(const float* __restrict__ pca,
                                                    _Float16* __restrict__ pcaP) {
    int i = blockIdx.x * 256 + threadIdx.x;
    int e = i & 15, lane = (i >> 4) & 31, tk = (i >> 9) & 31, tn = i >> 14;
    int k = tk * 32 + (lane >> 4) * 16 + e;
    int n = tn * 16 + (lane & 15);
    pcaP[i] = (_Float16)pca[(size_t)k * P_DIM + n];
}

// Pack centers^T [P=256 (k), K=4096 (n)] into B-fragment order:
//   idx = ((tn*8 + tk)*32 + lane)*16 + e ;  k(p) = tk*32 + (lane>>4)*16 + e ; n = tn*16 + (lane&15)
__global__ __launch_bounds__(256) void apb_pack_centers(const float* __restrict__ centers,
                                                        _Float16* __restrict__ centP) {
    int i = blockIdx.x * 256 + threadIdx.x;
    int e = i & 15, lane = (i >> 4) & 31, tk = (i >> 9) & 7, tn = i >> 12;
    int p = tk * 32 + (lane >> 4) * 16 + e;
    int c = tn * 16 + (lane & 15);
    centP[i] = (_Float16)centers[(size_t)c * P_DIM + p];
}

// c2[k] = sum(centers[k]^2)
__global__ __launch_bounds__(256) void apb_c2(const float* __restrict__ centers,
                                              float* __restrict__ c2) {
    __shared__ float red[256];
    int k = blockIdx.x, p = threadIdx.x;
    float v = centers[(size_t)k * P_DIM + p];
    red[p] = v * v;
    __syncthreads();
    for (int s = 128; s > 0; s >>= 1) {
        if (p < s) red[p] += red[p + s];
        __syncthreads();
    }
    if (p == 0) c2[k] = red[0];
}

// ---------------------------------------------------------------------------
// enc = ((flat-mean)*rstd) @ pca  -> f16 [B][256]
// 512 threads = 16 waves; block computes 16 rows x 256 cols.
// Whole 16x1024 normalized A tile staged once (32KB LDS), then 32 barrier-free WMMAs.
__global__ __launch_bounds__(512) void apb_encode(const float* __restrict__ actions,
                                                  const float* __restrict__ mean,
                                                  const float* __restrict__ rstd,
                                                  const _Float16* __restrict__ pcaP,
                                                  _Float16* __restrict__ enc16) {
    __shared__ __align__(16) _Float16 As[16][F_DIM + 8];   // pitch 1032: 16B-aligned, bank-spread
    const v16h* pcaPv = (const v16h*)pcaP;
    const int tid  = threadIdx.x;
    const int lane = tid & 31, w = tid >> 5;
    const int b0   = blockIdx.x * 16;
    const int n0   = w * 16;
    const int m    = lane & 15, h = lane >> 4, n = m;

    // stage: 2048 chunks of 8 halves; b128 loads, normalize, b128 LDS store
    for (int j = tid; j < 2048; j += 512) {
        int row = j >> 7, cc = (j & 127) * 8;
        const float* ap = actions + (size_t)(b0 + row) * F_DIM + cc;
        v4f x0 = *(const v4f*)(ap);
        v4f x1 = *(const v4f*)(ap + 4);
        v4f mu0 = *(const v4f*)(mean + cc);
        v4f mu1 = *(const v4f*)(mean + cc + 4);
        v4f r0 = *(const v4f*)(rstd + cc);
        v4f r1 = *(const v4f*)(rstd + cc + 4);
        v4f y0 = (x0 - mu0) * r0;
        v4f y1 = (x1 - mu1) * r1;
        v8h o;
#pragma unroll
        for (int q = 0; q < 4; ++q) { o[q] = (_Float16)y0[q]; o[4 + q] = (_Float16)y1[q]; }
        *(v8h*)&As[row][cc] = o;
    }
    __syncthreads();

    v8f acc = {};
#pragma unroll 4
    for (int kk = 0; kk < F_DIM / 32; ++kk) {
        v16h a  = apb_load_afrag(&As[m][kk * 32], h);
        v16h bf = pcaPv[(size_t)(w * 32 + kk) * 32 + lane];   // two b128 loads
        acc = __builtin_amdgcn_wmma_f32_16x16x32_f16(false, a, false, bf,
                                                     (short)0, acc, false, false);
    }
#pragma unroll
    for (int r = 0; r < 8; ++r) {
        int M = r + 8 * h;                  // C layout: lanes16-31 start at M=8
        enc16[(size_t)(b0 + M) * P_DIM + n0 + n] = (_Float16)acc[r];
    }
}

// e2[row] = sum(enc[row]^2); grid.y selects pred/target
__global__ __launch_bounds__(256) void apb_e2(const _Float16* __restrict__ encA,
                                              const _Float16* __restrict__ encB,
                                              float* __restrict__ e2A,
                                              float* __restrict__ e2B) {
    __shared__ float red[256];
    const _Float16* enc = blockIdx.y ? encB : encA;
    float* e2           = blockIdx.y ? e2B  : e2A;
    int row = blockIdx.x, p = threadIdx.x;
    float v = (float)enc[(size_t)row * P_DIM + p];
    red[p] = v * v;
    __syncthreads();
    for (int s = 128; s > 0; s >>= 1) {
        if (p < s) red[p] += red[p + s];
        __syncthreads();
    }
    if (p == 0) e2[row] = red[0];
}

// ---------------------------------------------------------------------------
// shared helpers for the two K-sweep kernels
#define APB_AE_DECL __shared__ __align__(16) _Float16 Ae[16][264]
#define APB_STAGE_AE(enc16, b0, tid)                                           \
    for (int j = (tid); j < 512; j += 256) {                                   \
        int row = j >> 5, cc = j & 31;                                         \
        *(v8h*)&Ae[row][cc * 8] =                                              \
            *(const v8h*)((enc16) + (size_t)((b0) + row) * P_DIM + cc * 8);    \
    }
// hoist the 8 invariant A fragments into registers (64 VGPRs, live across K loop)
#define APB_LOAD_AFRAGS(afr, m, h)                                             \
    v16h afr[8];                                                               \
    _Pragma("unroll")                                                          \
    for (int kk = 0; kk < P_DIM / 32; ++kk)                                    \
        afr[kk] = apb_load_afrag(&Ae[m][kk * 32], h);

// target pass: argmax over K of -max(e2+c2-2*enc.c, 0)
__global__ __launch_bounds__(256) void apb_target_argmax(const _Float16* __restrict__ enc16,
                                                         const float* __restrict__ e2,
                                                         const _Float16* __restrict__ centP,
                                                         const float* __restrict__ c2,
                                                         int* __restrict__ tids) {
    APB_AE_DECL;
    __shared__ float sE2[16];
    __shared__ float s_bm[8][16];
    __shared__ int   s_bi[8][16];
    const v16h* centPv = (const v16h*)centP;
    const int tid = threadIdx.x, lane = tid & 31, w = tid >> 5;
    const int b0  = blockIdx.x * 16;
    APB_STAGE_AE(enc16, b0, tid);
    if (tid < 16) sE2[tid] = e2[b0 + tid];
    __syncthreads();

    const int m = lane & 15, h = lane >> 4, n = m;
    APB_LOAD_AFRAGS(afr, m, h);

    float bm[8]; int bi[8];
#pragma unroll
    for (int r = 0; r < 8; ++r) { bm[r] = -INFINITY; bi[r] = 0x7fffffff; }

    for (int kt = 0; kt < K_DIM / 128; ++kt) {
        const int tn = kt * 8 + w;
        if (kt + 1 < K_DIM / 128)   // prefetch next tile's fragment stream
            __builtin_prefetch((const void*)(centPv + ((size_t)(tn + 8) * 8) * 32), 0, 0);
        v8f acc = {};
#pragma unroll
        for (int kk = 0; kk < P_DIM / 32; ++kk) {
            v16h bf = centPv[((size_t)tn * 8 + kk) * 32 + lane];  // two b128 loads
            acc = __builtin_amdgcn_wmma_f32_16x16x32_f16(false, afr[kk], false, bf,
                                                         (short)0, acc, false, false);
        }
        const int   ci  = tn * 16 + n;
        const float c2n = c2[ci];
#pragma unroll
        for (int r = 0; r < 8; ++r) {
            int M = r + 8 * h;
            float x = -fmaxf(sE2[M] + c2n - 2.0f * acc[r], 0.0f);
            if (x > bm[r]) { bm[r] = x; bi[r] = ci; }
        }
    }
    // reduce across 16 lanes of the half-wave (first-index tie break)
#pragma unroll
    for (int r = 0; r < 8; ++r) {
#pragma unroll
        for (int off = 1; off < 16; off <<= 1) {
            float om = __shfl_xor(bm[r], off);
            int   oi = __shfl_xor(bi[r], off);
            if (om > bm[r] || (om == bm[r] && oi < bi[r])) { bm[r] = om; bi[r] = oi; }
        }
    }
    if (m == 0) {
#pragma unroll
        for (int r = 0; r < 8; ++r) { s_bm[w][r + 8 * h] = bm[r]; s_bi[w][r + 8 * h] = bi[r]; }
    }
    __syncthreads();
    if (tid < 16) {
        float bestv = s_bm[0][tid]; int besti = s_bi[0][tid];
        for (int ww = 1; ww < 8; ++ww) {
            float om = s_bm[ww][tid]; int oi = s_bi[ww][tid];
            if (om > bestv || (om == bestv && oi < besti)) { bestv = om; besti = oi; }
        }
        tids[b0 + tid] = besti;
    }
}

// pred pass: online log-sum-exp + argmax + target-logit pick; atomic accumulate
__global__ __launch_bounds__(256) void apb_pred_pass(const _Float16* __restrict__ enc16,
                                                     const float* __restrict__ e2,
                                                     const _Float16* __restrict__ centP,
                                                     const float* __restrict__ c2,
                                                     const int* __restrict__ tids,
                                                     float* __restrict__ accum) {
    APB_AE_DECL;
    __shared__ float sE2[16];
    __shared__ int   sTid[16];
    __shared__ float s_m[8][16], s_s[8][16], s_tl[8][16];
    __shared__ int   s_ai[8][16];
    const v16h* centPv = (const v16h*)centP;
    const int tid = threadIdx.x, lane = tid & 31, w = tid >> 5;
    const int b0  = blockIdx.x * 16;
    APB_STAGE_AE(enc16, b0, tid);
    if (tid < 16) { sE2[tid] = e2[b0 + tid]; sTid[tid] = tids[b0 + tid]; }
    __syncthreads();

    const int m = lane & 15, h = lane >> 4, n = m;
    APB_LOAD_AFRAGS(afr, m, h);

    float rm[8], rs[8], rtl[8]; int rai[8];
#pragma unroll
    for (int r = 0; r < 8; ++r) { rm[r] = -INFINITY; rs[r] = 0.0f; rtl[r] = -INFINITY; rai[r] = 0x7fffffff; }

    for (int kt = 0; kt < K_DIM / 128; ++kt) {
        const int tn = kt * 8 + w;
        if (kt + 1 < K_DIM / 128)
            __builtin_prefetch((const void*)(centPv + ((size_t)(tn + 8) * 8) * 32), 0, 0);
        v8f acc = {};
#pragma unroll
        for (int kk = 0; kk < P_DIM / 32; ++kk) {
            v16h bf = centPv[((size_t)tn * 8 + kk) * 32 + lane];
            acc = __builtin_amdgcn_wmma_f32_16x16x32_f16(false, afr[kk], false, bf,
                                                         (short)0, acc, false, false);
        }
        const int   ci  = tn * 16 + n;
        const float c2n = c2[ci];
#pragma unroll
        for (int r = 0; r < 8; ++r) {
            int M = r + 8 * h;
            float x = -fmaxf(sE2[M] + c2n - 2.0f * acc[r], 0.0f);
            if (x > rm[r]) {                       // new running max
                rs[r] = rs[r] * __expf(rm[r] - x) + 1.0f;
                rm[r] = x; rai[r] = ci;
            } else {
                rs[r] += __expf(x - rm[r]);
            }
            if (ci == sTid[M]) rtl[r] = x;
        }
    }
    // combine partial (max,sum,argmax,target-logit) across 16-lane group
#pragma unroll
    for (int r = 0; r < 8; ++r) {
#pragma unroll
        for (int off = 1; off < 16; off <<= 1) {
            float om  = __shfl_xor(rm[r],  off);
            float os  = __shfl_xor(rs[r],  off);
            float otl = __shfl_xor(rtl[r], off);
            int   oi  = __shfl_xor(rai[r], off);
            float Mx = fmaxf(rm[r], om);
            float S  = rs[r] * __expf(rm[r] - Mx) + os * __expf(om - Mx);
            if (om > rm[r] || (om == rm[r] && oi < rai[r])) rai[r] = oi;
            rm[r] = Mx; rs[r] = S; rtl[r] = fmaxf(rtl[r], otl);
        }
    }
    if (m == 0) {
#pragma unroll
        for (int r = 0; r < 8; ++r) {
            int M = r + 8 * h;
            s_m[w][M] = rm[r]; s_s[w][M] = rs[r]; s_tl[w][M] = rtl[r]; s_ai[w][M] = rai[r];
        }
    }
    __syncthreads();
    if (tid < 16) {
        float Mx = s_m[0][tid], S = s_s[0][tid], TL = s_tl[0][tid];
        int   AI = s_ai[0][tid];
        for (int ww = 1; ww < 8; ++ww) {
            float om = s_m[ww][tid], os = s_s[ww][tid], otl = s_tl[ww][tid];
            int   oi = s_ai[ww][tid];
            float M2 = fmaxf(Mx, om);
            S = S * __expf(Mx - M2) + os * __expf(om - M2);
            if (om > Mx || (om == Mx && oi < AI)) AI = oi;
            Mx = M2; TL = fmaxf(TL, otl);
        }
        float logp_t = TL - Mx - __logf(S);
        atomicAdd(&accum[0], -logp_t);
        atomicAdd(&accum[1], (AI == sTid[tid]) ? 1.0f : 0.0f);
    }
}

__global__ __launch_bounds__(64) void apb_finalize(const float* __restrict__ accum,
                                                   float* __restrict__ out) {
    if (threadIdx.x == 0 && blockIdx.x == 0) {
        out[0] = accum[0] * (1.0f / (float)B_DIM);
        out[1] = accum[1] * (1.0f / (float)B_DIM);
    }
}

// ---------------------------------------------------------------------------
extern "C" void kernel_launch(void* const* d_in, const int* in_sizes, int n_in,
                              void* d_out, int out_size, void* d_ws, size_t ws_size,
                              hipStream_t stream) {
    (void)in_sizes; (void)n_in; (void)out_size; (void)ws_size;
    const float* pred    = (const float*)d_in[0];   // [B,T,D] = [B,1024]
    const float* target  = (const float*)d_in[1];
    const float* centers = (const float*)d_in[2];   // [K,256]
    const float* mean    = (const float*)d_in[3];   // [1024]
    const float* stdv    = (const float*)d_in[4];   // [1024]
    const float* pca     = (const float*)d_in[5];   // [1024,256]
    float* out = (float*)d_out;

    // workspace carve-up (~19 MB total)
    char* ws = (char*)d_ws;
    size_t off = 0;
    _Float16* encp16 = (_Float16*)(ws + off); off += (size_t)B_DIM * P_DIM * 2;  // 8 MB
    _Float16* enct16 = (_Float16*)(ws + off); off += (size_t)B_DIM * P_DIM * 2;  // 8 MB
    _Float16* pcaP   = (_Float16*)(ws + off); off += (size_t)F_DIM * P_DIM * 2;  // 512 KB
    _Float16* centP  = (_Float16*)(ws + off); off += (size_t)P_DIM * K_DIM * 2;  // 2 MB
    float*    c2     = (float*)   (ws + off); off += (size_t)K_DIM * 4;
    float*    e2p    = (float*)   (ws + off); off += (size_t)B_DIM * 4;
    float*    e2t    = (float*)   (ws + off); off += (size_t)B_DIM * 4;
    int*      tids   = (int*)     (ws + off); off += (size_t)B_DIM * 4;
    float*    rstd   = (float*)   (ws + off); off += (size_t)F_DIM * 4;
    float*    accum  = (float*)   (ws + off); off += 256;

    apb_zero_accum<<<1, 256, 0, stream>>>(accum);
    apb_rstd<<<F_DIM / 256, 256, 0, stream>>>(stdv, rstd);
    apb_pack_pca<<<(F_DIM * P_DIM) / 256, 256, 0, stream>>>(pca, pcaP);
    apb_pack_centers<<<(P_DIM * K_DIM) / 256, 256, 0, stream>>>(centers, centP);
    apb_c2<<<K_DIM, 256, 0, stream>>>(centers, c2);

    apb_encode<<<B_DIM / 16, 512, 0, stream>>>(pred,   mean, rstd, pcaP, encp16);
    apb_encode<<<B_DIM / 16, 512, 0, stream>>>(target, mean, rstd, pcaP, enct16);

    apb_e2<<<dim3(B_DIM, 2), 256, 0, stream>>>(encp16, enct16, e2p, e2t);

    apb_target_argmax<<<B_DIM / 16, 256, 0, stream>>>(enct16, e2t, centP, c2, tids);
    apb_pred_pass<<<B_DIM / 16, 256, 0, stream>>>(encp16, e2p, centP, c2, tids, accum);

    apb_finalize<<<1, 64, 0, stream>>>(accum, out);
}
